// LongformerClass_81621558493267
// MI455X (gfx1250) — matrix operand order repaired and checked by reading, not compiled, not run on hardware
//
#include <hip/hip_runtime.h>

// ---------------- config (longformer-base-4096) ----------------
#define S_LEN 4096
#define DM    768
#define NH    12
#define DHD   64
#define FFD   3072
#define NL    12
#define WW    256   // one-sided window; key window = 3*WW
#define NCHK  (S_LEN / WW)

typedef unsigned short u16;
typedef unsigned int   u32;
typedef __attribute__((ext_vector_type(16))) __bf16 bf16x16;
typedef __attribute__((ext_vector_type(8)))  float  f32x8;
typedef __attribute__((ext_vector_type(4)))  u32    u32x4;
typedef __attribute__((ext_vector_type(8)))  int    i32x8;
typedef __attribute__((ext_vector_type(4)))  int    i32x4;

union Frag { bf16x16 v; uint4 q[2]; u16 us[16]; };

__device__ __forceinline__ u16 f2bf(float f) {
  unsigned u = __float_as_uint(f);
  u += 0x7fffu + ((u >> 16) & 1u);          // round-to-nearest-even
  return (u16)(u >> 16);
}

__device__ __forceinline__ f32x8 wmma_bf16(Frag a, Frag b, f32x8 c) {
  // D = A(16x32 bf16) * B(32x16 bf16) + C(16x16 f32)
  return __builtin_amdgcn_wmma_f32_16x16x32_bf16(
      false, a.v, false, b.v, (short)0, c, false, false);
}

// low 32 bits of a flat pointer to __shared__ = LDS byte offset
__device__ __forceinline__ unsigned lds_off(const void* p) {
  return (unsigned)(unsigned long long)(uintptr_t)p;
}

// ---------------- Tensor Data Mover: 2D bf16 tile -> LDS ----------------
// Loads tile_d1 rows x tile_d0 elems (2B each), row stride = stride0 elems.
// LDS padding: +4 DWORDs after every 32 DWORDs (one 64-elem row) -> 144B pitch.
__device__ __forceinline__ void tdm_load_2d(
    unsigned lds_addr, const void* gptr,
    u32 tensor_d0, u32 tensor_d1, u32 tile_d0, u32 tile_d1,
    unsigned long long stride0) {
  const unsigned long long ga = (unsigned long long)(uintptr_t)gptr;
  u32x4 g0;
  g0[0] = 1u;                                  // count=1 (valid), user descriptor
  g0[1] = lds_addr;                            // LDS byte address
  g0[2] = (u32)ga;                             // global_addr[31:0]
  g0[3] = (u32)(ga >> 32) | (2u << 30);        // global_addr[56:32] | type=2
  i32x8 g1;
  // data_size=1 (2B), pad_enable, pad_interval code 4 (32 DW), pad_amount code 3 (4 DW)
  g1[0] = (int)((1u << 16) | (1u << 20) | (4u << 22) | (3u << 25));
  g1[1] = (int)((tensor_d0 & 0xFFFFu) << 16);                       // tensor_dim0 lo
  g1[2] = (int)((tensor_d0 >> 16) | ((tensor_d1 & 0xFFFFu) << 16)); // dim0 hi | dim1 lo
  g1[3] = (int)((tensor_d1 >> 16) | (tile_d0 << 16));               // dim1 hi | tile_dim0
  g1[4] = (int)(tile_d1 & 0xFFFFu);                                 // tile_dim1 (tile_dim2=0)
  g1[5] = (int)(u32)stride0;                                        // tensor_dim0_stride lo
  g1[6] = (int)((u32)(stride0 >> 32) & 0xFFFFu);                    // stride hi (dim1_stride=0)
  g1[7] = 0;
  const i32x4 z4 = {0, 0, 0, 0};
  const i32x8 z8 = {0, 0, 0, 0, 0, 0, 0, 0};
  // 6-arg toolchain flavor: (g0, g1, g2, g3, extra_group, cpol)
  __builtin_amdgcn_tensor_load_to_lds(g0, g1, z4, z4, z8, 0);
}

// ---------------- position ids (cumsum of mask, padding_idx=1) ----------------
__global__ void lf_pos_scan(const int* __restrict__ mask, int* __restrict__ pos) {
  if (threadIdx.x == 0 && blockIdx.x == 0) {
    int run = 0;
    for (int i = 0; i < S_LEN; ++i) { run += mask[i]; pos[i] = run * mask[i] + 1; }
  }
}

// ---------------- shared LN helper: 256-thread block sum ----------------
__device__ __forceinline__ float block_sum256(float v, float* sb, float* slot) {
#pragma unroll
  for (int o = 16; o; o >>= 1) v += __shfl_xor(v, o, 32);
  __syncthreads();
  if ((threadIdx.x & 31) == 0) sb[threadIdx.x >> 5] = v;
  __syncthreads();
  if (threadIdx.x == 0) { float a = 0.f; for (int i = 0; i < 8; ++i) a += sb[i]; *slot = a; }
  __syncthreads();
  return *slot;
}

// ---------------- embedding + LN -> x fp32 and x bf16 ----------------
__global__ __launch_bounds__(256) void lf_embed_ln(
    const int* __restrict__ ids, const int* __restrict__ pos,
    const float* __restrict__ we, const float* __restrict__ pe,
    const float* __restrict__ te,
    const float* __restrict__ gma, const float* __restrict__ bta,
    float* __restrict__ xf, u16* __restrict__ xb) {
  __shared__ float sb[8];
  __shared__ float stat[2];
  const int s = blockIdx.x, tid = threadIdx.x;
  const int id = ids[s], p = pos[s];
  float e[3];
#pragma unroll
  for (int u = 0; u < 3; ++u) {
    int j = tid + u * 256;
    e[u] = we[(size_t)id * DM + j] + pe[(size_t)p * DM + j] + te[j];
  }
  const float mean = block_sum256(e[0] + e[1] + e[2], sb, &stat[0]) * (1.f / DM);
  float vq = 0.f;
#pragma unroll
  for (int u = 0; u < 3; ++u) { float d = e[u] - mean; vq += d * d; }
  const float var = block_sum256(vq, sb, &stat[1]) * (1.f / DM);
  const float inv = rsqrtf(var + 1e-5f);
#pragma unroll
  for (int u = 0; u < 3; ++u) {
    int j = tid + u * 256;
    float o_ = (e[u] - mean) * inv * gma[j] + bta[j];
    xf[(size_t)s * DM + j] = o_;
    xb[(size_t)s * DM + j] = f2bf(o_);
  }
}

// ---------------- residual + LN: x = LN(x + y) ----------------
__global__ __launch_bounds__(256) void lf_residual_ln(
    float* __restrict__ xf, u16* __restrict__ xb, const float* __restrict__ y,
    const float* __restrict__ gma, const float* __restrict__ bta) {
  __shared__ float sb[8];
  __shared__ float stat[2];
  const int s = blockIdx.x, tid = threadIdx.x;
  float e[3];
#pragma unroll
  for (int u = 0; u < 3; ++u) {
    int j = tid + u * 256;
    e[u] = xf[(size_t)s * DM + j] + y[(size_t)s * DM + j];
  }
  const float mean = block_sum256(e[0] + e[1] + e[2], sb, &stat[0]) * (1.f / DM);
  float vq = 0.f;
#pragma unroll
  for (int u = 0; u < 3; ++u) { float d = e[u] - mean; vq += d * d; }
  const float var = block_sum256(vq, sb, &stat[1]) * (1.f / DM);
  const float inv = rsqrtf(var + 1e-5f);
#pragma unroll
  for (int u = 0; u < 3; ++u) {
    int j = tid + u * 256;
    float o_ = (e[u] - mean) * inv * gma[j] + bta[j];
    xf[(size_t)s * DM + j] = o_;
    xb[(size_t)s * DM + j] = f2bf(o_);
  }
}

// ---------------- weight fp32[K,N] -> bf16 transposed [N,K] ----------------
__global__ __launch_bounds__(256) void lf_w_to_bf16t(
    const float* __restrict__ src, u16* __restrict__ dst, int K, int N) {
  __shared__ u16 t[32][33];
  const int n0 = blockIdx.x * 32, k0 = blockIdx.y * 32;
  const int tx = threadIdx.x, ty = threadIdx.y;
  for (int r = ty; r < 32; r += 8)
    t[r][tx] = f2bf(src[(size_t)(k0 + r) * N + n0 + tx]);
  __syncthreads();
  for (int r = ty; r < 32; r += 8)
    dst[(size_t)(n0 + r) * K + k0 + tx] = t[tx][r];
}

// ---------------- bf16 [S,DM] -> [DM,S] (per-head V transpose) ----------------
__global__ __launch_bounds__(256) void lf_v_transpose(
    const u16* __restrict__ src, u16* __restrict__ dst) {
  __shared__ u16 t[32][33];
  const int c0 = blockIdx.x * 32, r0 = blockIdx.y * 32;
  const int tx = threadIdx.x, ty = threadIdx.y;
  for (int r = ty; r < 32; r += 8)
    t[r][tx] = src[(size_t)(r0 + r) * DM + c0 + tx];
  __syncthreads();
  for (int r = ty; r < 32; r += 8)
    dst[(size_t)(c0 + r) * S_LEN + r0 + tx] = t[tx][r];
}

// ---------------- WMMA GEMM with TDM-staged weight panel ----------------
// C[M,N] = A[M,K]bf16 x Wt[N,K]bf16 + bias
// Block: 8 waves, 128 rows x 64 cols. Weight panel (64 x 64 per stage) is
// DMA'd into LDS by the Tensor Data Mover (double-buffered, TENSORcnt),
// with TDM LDS padding (144B row pitch) for conflict-free ds_load_b128.
// MODE 0: fp32 out; MODE 1: bf16 out; MODE 2: GELU(exact) -> bf16 out
#define KSTEP   64
#define BPITCHE 72   // staged row pitch in bf16 elements (144 bytes)

template <int MODE>
__global__ __launch_bounds__(256) void lf_gemm_bf16(
    const u16* __restrict__ A, const u16* __restrict__ Wt,
    const float* __restrict__ bias, void* __restrict__ outv,
    int M, int N, int K) {
  __shared__ __align__(16) u16 bstage[2][64 * BPITCHE];   // 2 x 9216 B
  const int wave = threadIdx.x >> 5;
  const int lane = threadIdx.x & 31;
  const int lo = lane & 15, hi = lane >> 4;
  const int m0 = blockIdx.x * 128 + wave * 16;
  const int n0 = blockIdx.y * 64;
  const u16* arow = A + (size_t)(m0 + lo) * K;
  const int nstage = K / KSTEP;

  if (wave == 0) {
    tdm_load_2d(lds_off(&bstage[0][0]), Wt + (size_t)n0 * K,
                (u32)K, (u32)N, KSTEP, 64, (unsigned long long)K);
  }

  f32x8 acc[4] = {};
  for (int ks = 0; ks < nstage; ++ks) {
    const int cur = ks & 1;
    if (wave == 0) {
      if (ks + 1 < nstage) {
        tdm_load_2d(lds_off(&bstage[cur ^ 1][0]),
                    Wt + (size_t)n0 * K + (size_t)(ks + 1) * KSTEP,
                    (u32)K, (u32)N, KSTEP, 64, (unsigned long long)K);
        __builtin_amdgcn_s_wait_tensorcnt(1);   // stage `cur` complete (in-order)
      } else {
        __builtin_amdgcn_s_wait_tensorcnt(0);
      }
    }
    __syncthreads();                            // publish stage to all waves
    const u16* bbase = &bstage[cur][0];
#pragma unroll
    for (int half = 0; half < 2; ++half) {
      const int kk = ks * KSTEP + half * 32;
      Frag a;                                   // A 16x32: k = hi*16 + i
      a.q[0] = *(const uint4*)(arow + kk + hi * 16);
      a.q[1] = *(const uint4*)(arow + kk + hi * 16 + 8);
#pragma unroll
      for (int nt = 0; nt < 4; ++nt) {          // B 32x16: n = lo, k = hi*16 + i
        const u16* brow = bbase + (size_t)(nt * 16 + lo) * BPITCHE + half * 32 + hi * 16;
        Frag b;
        b.q[0] = *(const uint4*)(brow);
        b.q[1] = *(const uint4*)(brow + 8);
        acc[nt] = wmma_bf16(a, b, acc[nt]);
      }
    }
    __syncthreads();                            // readers done before reuse of `cur`
  }
#pragma unroll
  for (int nt = 0; nt < 4; ++nt) {
    const int nn = n0 + nt * 16 + lo;
    const float bi = bias[nn];
#pragma unroll
    for (int v = 0; v < 8; ++v) {               // D: m = v + hi*8, n = lo
      const int m = m0 + v + hi * 8;
      float val = acc[nt][v] + bi;
      if (MODE == 0) {
        ((float*)outv)[(size_t)m * N + nn] = val;
      } else if (MODE == 1) {
        ((u16*)outv)[(size_t)m * N + nn] = f2bf(val);
      } else {
        float g = 0.5f * val * (1.0f + erff(val * 0.70710678118654752f));
        ((u16*)outv)[(size_t)m * N + nn] = f2bf(g);
      }
    }
  }
}

// ---------------- sliding-window attention ----------------
// grid (WW/16 row-tiles, NCHK chunks, NH heads), 1 wave per block
__global__ __launch_bounds__(32) void lf_attn(
    const u16* __restrict__ Q, const u16* __restrict__ Kmat,
    const u16* __restrict__ Vt, const int* __restrict__ msk,
    u16* __restrict__ O) {
  __shared__ float sc[16 * 3 * WW];             // 16 x 768 scores -> probs (48 KB)
  const int lane = threadIdx.x;
  const int lo = lane & 15, hi = lane >> 4;
  const int t = blockIdx.x, c = blockIdx.y, h = blockIdx.z;
  const int r0 = c * WW + t * 16;

  Frag qa[2];
  {
    const u16* qp = Q + (size_t)(r0 + lo) * DM + h * DHD;
#pragma unroll
    for (int f = 0; f < 2; ++f) {
      qa[f].q[0] = *(const uint4*)(qp + f * 32 + hi * 16);
      qa[f].q[1] = *(const uint4*)(qp + f * 32 + hi * 16 + 8);
    }
  }

  // phase 1: S = Q K^T (contract over dh=64 -> 2 WMMAs per 16-key tile)
  for (int j = 0; j < (3 * WW) / 16; ++j) {
    const int kj = j * 16 + lo;                 // window key index for this lane
    const int g = c * WW - WW + kj;             // global key index
    const bool kvalid = (g >= 0) && (g < S_LEN) && (msk[(g >= 0 && g < S_LEN) ? g : 0] != 0);
    Frag kf[2];
    if (kvalid) {
      const u16* kp = Kmat + (size_t)g * DM + h * DHD + hi * 16;
      kf[0].q[0] = *(const uint4*)(kp);
      kf[0].q[1] = *(const uint4*)(kp + 8);
      kf[1].q[0] = *(const uint4*)(kp + 32);
      kf[1].q[1] = *(const uint4*)(kp + 40);
    } else {
      const uint4 z = make_uint4(0u, 0u, 0u, 0u);
      kf[0].q[0] = z; kf[0].q[1] = z; kf[1].q[0] = z; kf[1].q[1] = z;
    }
    f32x8 s = {};
    s = wmma_bf16(qa[0], kf[0], s);
    s = wmma_bf16(qa[1], kf[1], s);
#pragma unroll
    for (int v = 0; v < 8; ++v) {
      const int qi = t * 16 + v + hi * 8;       // query index within chunk
      const bool allowed = kvalid && (kj >= qi) && (kj <= qi + 2 * WW);
      sc[(v + hi * 8) * (3 * WW) + kj] = allowed ? s[v] * 0.125f : -1e30f;
    }
  }
  __syncthreads();

  // phase 2: softmax, 2 lanes per row
  {
    const int r = lane >> 1, hf = lane & 1;
    float* row = sc + r * (3 * WW) + hf * 384;
    float mx = -1e30f;
    for (int i = 0; i < 384; ++i) mx = fmaxf(mx, row[i]);
    mx = fmaxf(mx, __shfl_xor(mx, 1, 32));
    float sum = 0.f;
    for (int i = 0; i < 384; ++i) sum += __expf(row[i] - mx);
    sum += __shfl_xor(sum, 1, 32);
    const float inv = 1.f / fmaxf(sum, 1e-30f);
    for (int i = 0; i < 384; ++i) row[i] = __expf(row[i] - mx) * inv;
  }
  __syncthreads();

  // phase 3: O = P x V  (A from LDS probs, B from V^T [DM][S])
  f32x8 o[4] = {};
  for (int kt = 0; kt < (3 * WW) / 32; ++kt) {
    Frag pa;                                    // A: k = hi*16 + i
    const float* pr = sc + lo * (3 * WW) + kt * 32 + hi * 16;
    const float4 p0 = *(const float4*)(pr);
    const float4 p1 = *(const float4*)(pr + 4);
    const float4 p2 = *(const float4*)(pr + 8);
    const float4 p3 = *(const float4*)(pr + 12);
    pa.us[0] = f2bf(p0.x); pa.us[1] = f2bf(p0.y); pa.us[2]  = f2bf(p0.z); pa.us[3]  = f2bf(p0.w);
    pa.us[4] = f2bf(p1.x); pa.us[5] = f2bf(p1.y); pa.us[6]  = f2bf(p1.z); pa.us[7]  = f2bf(p1.w);
    pa.us[8] = f2bf(p2.x); pa.us[9] = f2bf(p2.y); pa.us[10] = f2bf(p2.z); pa.us[11] = f2bf(p2.w);
    pa.us[12]= f2bf(p3.x); pa.us[13]= f2bf(p3.y); pa.us[14] = f2bf(p3.z); pa.us[15] = f2bf(p3.w);
    const int gbase = c * WW - WW + kt * 32 + hi * 16;
#pragma unroll
    for (int nt = 0; nt < 4; ++nt) {            // B: n = dh = nt*16+lo, k = key
      const u16* vp = Vt + (size_t)(h * DHD + nt * 16 + lo) * S_LEN + gbase;
      Frag vf;
      if (gbase >= 0 && gbase + 16 <= S_LEN) {
        vf.q[0] = *(const uint4*)(vp);
        vf.q[1] = *(const uint4*)(vp + 8);
      } else {
#pragma unroll
        for (int i = 0; i < 16; ++i) {
          const int g = gbase + i;
          vf.us[i] = (g >= 0 && g < S_LEN) ? vp[i] : (u16)0;
        }
      }
      o[nt] = wmma_bf16(pa, vf, o[nt]);
    }
  }
#pragma unroll
  for (int nt = 0; nt < 4; ++nt)
#pragma unroll
    for (int v = 0; v < 8; ++v)
      O[(size_t)(r0 + v + hi * 8) * DM + h * DHD + nt * 16 + lo] = f2bf(o[nt][v]);
}

// ---------------- pooler + classifier ----------------
__global__ __launch_bounds__(256) void lf_pool_cls(
    const float* __restrict__ xf, const float* __restrict__ wp,
    const float* __restrict__ bp, const float* __restrict__ wc,
    const float* __restrict__ bc, float* __restrict__ out) {
  __shared__ float x0[DM];
  __shared__ float pooled[DM];
  const int tid = threadIdx.x;
#pragma unroll
  for (int u = 0; u < 3; ++u) x0[tid + u * 256] = xf[tid + u * 256];
  __syncthreads();
  for (int n = tid; n < DM; n += 256) {
    float a = bp[n];
    for (int k = 0; k < DM; ++k) a += x0[k] * wp[(size_t)k * DM + n];
    pooled[n] = tanhf(a);
  }
  __syncthreads();
  if (tid < 6) {
    float a = bc[tid];
    for (int k = 0; k < DM; ++k) a += pooled[k] * wc[(size_t)k * 6 + tid];
    out[tid] = a;
  }
}

// ---------------- host launcher ----------------
extern "C" void kernel_launch(void* const* d_in, const int* in_sizes, int n_in,
                              void* d_out, int out_size, void* d_ws, size_t ws_size,
                              hipStream_t stream) {
  (void)in_sizes; (void)n_in; (void)out_size; (void)ws_size;
  const int*   ids   = (const int*)  d_in[0];
  const int*   mask  = (const int*)  d_in[1];
  const float* wemb  = (const float*)d_in[2];
  const float* pemb  = (const float*)d_in[3];
  const float* temb  = (const float*)d_in[4];
  const float* lnEs  = (const float*)d_in[5];
  const float* lnEb  = (const float*)d_in[6];
  const float* wq = (const float*)d_in[7];  const float* bq = (const float*)d_in[8];
  const float* wk = (const float*)d_in[9];  const float* bk = (const float*)d_in[10];
  const float* wv = (const float*)d_in[11]; const float* bv = (const float*)d_in[12];
  const float* wo = (const float*)d_in[13]; const float* bo = (const float*)d_in[14];
  const float* ln1s = (const float*)d_in[15]; const float* ln1b = (const float*)d_in[16];
  const float* w1 = (const float*)d_in[17]; const float* b1 = (const float*)d_in[18];
  const float* w2 = (const float*)d_in[19]; const float* b2 = (const float*)d_in[20];
  const float* ln2s = (const float*)d_in[21]; const float* ln2b = (const float*)d_in[22];
  const float* wpool = (const float*)d_in[23]; const float* bpool = (const float*)d_in[24];
  const float* wcls  = (const float*)d_in[25]; const float* bcls  = (const float*)d_in[26];

  char* wsp = (char*)d_ws;
  auto carve = [&](size_t bytes) -> void* {
    void* p = (void*)wsp;
    wsp += (bytes + 255) & ~(size_t)255;
    return p;
  };
  int*   pos  = (int*)  carve((size_t)S_LEN * 4);
  float* xf   = (float*)carve((size_t)S_LEN * DM * 4);
  u16*   xb   = (u16*)  carve((size_t)S_LEN * DM * 2);
  u16*   qb   = (u16*)  carve((size_t)S_LEN * DM * 2);
  u16*   kbuf = (u16*)  carve((size_t)S_LEN * DM * 2);
  u16*   vbuf = (u16*)  carve((size_t)S_LEN * DM * 2);
  u16*   vT   = (u16*)  carve((size_t)S_LEN * DM * 2);
  u16*   ab   = (u16*)  carve((size_t)S_LEN * DM * 2);
  float* ybuf = (float*)carve((size_t)S_LEN * DM * 4);
  u16*   hb   = (u16*)  carve((size_t)S_LEN * FFD * 2);
  u16*   wstg = (u16*)  carve((size_t)DM * FFD * 2);

  lf_pos_scan<<<1, 1, 0, stream>>>(mask, pos);
  lf_embed_ln<<<S_LEN, 256, 0, stream>>>(ids, pos, wemb, pemb, temb, lnEs, lnEb, xf, xb);

  const dim3 tb(32, 8);
  const dim3 gDD(DM / 32, DM / 32);
  const dim3 gDF(FFD / 32, DM / 32);          // w1: K=DM rows, N=FFD cols
  const dim3 gFD(DM / 32, FFD / 32);          // w2: K=FFD rows, N=DM cols
  const dim3 gemmD(S_LEN / 128, DM / 64);
  const dim3 gemmF(S_LEN / 128, FFD / 64);

  for (int l = 0; l < NL; ++l) {
    const size_t oDD = (size_t)l * DM * DM;
    const size_t oD  = (size_t)l * DM;
    const size_t oDF = (size_t)l * DM * FFD;
    const size_t oF  = (size_t)l * FFD;

    lf_w_to_bf16t<<<gDD, tb, 0, stream>>>(wq + oDD, wstg, DM, DM);
    lf_gemm_bf16<1><<<gemmD, 256, 0, stream>>>(xb, wstg, bq + oD, qb, S_LEN, DM, DM);
    lf_w_to_bf16t<<<gDD, tb, 0, stream>>>(wk + oDD, wstg, DM, DM);
    lf_gemm_bf16<1><<<gemmD, 256, 0, stream>>>(xb, wstg, bk + oD, kbuf, S_LEN, DM, DM);
    lf_w_to_bf16t<<<gDD, tb, 0, stream>>>(wv + oDD, wstg, DM, DM);
    lf_gemm_bf16<1><<<gemmD, 256, 0, stream>>>(xb, wstg, bv + oD, vbuf, S_LEN, DM, DM);

    lf_v_transpose<<<dim3(DM / 32, S_LEN / 32), tb, 0, stream>>>(vbuf, vT);
    lf_attn<<<dim3(WW / 16, NCHK, NH), 32, 0, stream>>>(qb, kbuf, vT, mask, ab);

    lf_w_to_bf16t<<<gDD, tb, 0, stream>>>(wo + oDD, wstg, DM, DM);
    lf_gemm_bf16<0><<<gemmD, 256, 0, stream>>>(ab, wstg, bo + oD, ybuf, S_LEN, DM, DM);
    lf_residual_ln<<<S_LEN, 256, 0, stream>>>(xf, xb, ybuf, ln1s + oD, ln1b + oD);

    lf_w_to_bf16t<<<gDF, tb, 0, stream>>>(w1 + oDF, wstg, DM, FFD);
    lf_gemm_bf16<2><<<gemmF, 256, 0, stream>>>(xb, wstg, b1 + oF, hb, S_LEN, FFD, DM);
    lf_w_to_bf16t<<<gFD, tb, 0, stream>>>(w2 + oDF, wstg, FFD, DM);
    lf_gemm_bf16<0><<<gemmD, 256, 0, stream>>>(hb, wstg, b2 + oD, ybuf, S_LEN, DM, FFD);
    lf_residual_ln<<<S_LEN, 256, 0, stream>>>(xf, xb, ybuf, ln2s + oD, ln2b + oD);
  }

  lf_pool_cls<<<1, 256, 0, stream>>>(xf, wpool, bpool, wcls, bcls, (float*)d_out);
}
